// GPTBase_61795989455495
// MI455X (gfx1250) — matrix operand order repaired and checked by reading, not compiled, not run on hardware
//
#include <hip/hip_runtime.h>
#include <cstdint>
#include <cstddef>

// ---------------------------------------------------------------------------
// GPT forward (B=2,T=1024,V=50257,D=768,H=12,L=6,W=256) for gfx1250 (MI455X).
// All matmuls via v_wmma_f32_16x16x32_bf16 (wave32 WMMA, f32 accumulate).
// Weights converted f32->bf16 and transposed to [N,K]; GEMM staging uses the
// CDNA5 async global->LDS path (ASYNCcnt) when the builtins are available.
// ---------------------------------------------------------------------------

#define DEV __device__ __forceinline__

#if defined(__AMDGCN__) && __has_builtin(__builtin_amdgcn_global_load_async_to_lds_b64) && __has_builtin(__builtin_amdgcn_s_wait_asynccnt)
#define USE_ASYNC_LDS 1
#else
#define USE_ASYNC_LDS 0
#endif

typedef __attribute__((ext_vector_type(16))) __bf16 v16bf;
typedef __attribute__((ext_vector_type(8)))  float  v8f;

#if USE_ASYNC_LDS
// Parameter types per clang diagnostic: b64 variant takes
// (int2 AS1*, int2 AS3*, int imm_offset, int cpol).
typedef int v2i __attribute__((vector_size(8)));
typedef __attribute__((address_space(1))) v2i* gas_b64_ptr;
typedef __attribute__((address_space(3))) v2i* las_b64_ptr;
#endif

constexpr int Bb = 2, Tt = 1024, Vv = 50257, Dd = 768, Hh = 12, Ll = 6, Ww = 256;
constexpr int BT = Bb * Tt;          // 2048 rows
constexpr int D3 = 3 * Dd;           // 2304
constexpr int D4 = 4 * Dd;           // 3072
constexpr int LDA = 36;              // GEMM LDS row stride (72B: 8B-aligned rows)

DEV unsigned short f2bf(float f) {
  unsigned u = __float_as_uint(f);
  u += 0x7FFFu + ((u >> 16) & 1u);   // round-to-nearest-even
  return (unsigned short)(u >> 16);
}

union FragU { v16bf v; unsigned int u[8]; unsigned short s[16]; };

DEV v8f v8f_zero() {
  v8f v;
#pragma unroll
  for (int i = 0; i < 8; ++i) v[i] = 0.f;
  return v;
}

DEV void async_wait_all() {
#if USE_ASYNC_LDS
  __builtin_amdgcn_s_wait_asynccnt(0);
#endif
}

DEV void async_copy_b64(const unsigned short* src, unsigned short* dst_lds) {
#if USE_ASYNC_LDS
  __builtin_amdgcn_global_load_async_to_lds_b64(
      (gas_b64_ptr)(src), (las_b64_ptr)(dst_lds), 0, 0);
#else
  const uint2 v = *(const uint2*)src;
  *(unsigned int*)(dst_lds)     = v.x;
  *(unsigned int*)(dst_lds + 2) = v.y;
#endif
}

// A fragment: 16x32 bf16 tile at `base` (row-major M x K, lda even).
// lanes 0-15 -> M=lane (K-half 0), lanes 16-31 -> same M (K-half 1).
// VGPR j<4: K = kh*8 + 2j ; j>=4: K = 16 + kh*8 + 2(j-4). Pairs contiguous.
DEV v16bf load_a_frag(const unsigned short* base, int lda, int lane) {
  const int m = lane & 15, kh = lane >> 4;
  FragU r;
#pragma unroll
  for (int j = 0; j < 8; ++j) {
    const int k0 = (j < 4) ? (kh * 8 + 2 * j) : (16 + kh * 8 + 2 * (j - 4));
    r.u[j] = *(const unsigned int*)(base + m * lda + k0);
  }
  return r.v;
}

// B fragment from an N-major tile: base -> [16 n][K] row-major (ldb even).
// Lanes 0-15: N=lane (K-half 0); lanes 16-31: N=lane-16 (K-half 1).
// VGPR j<4: K = kh*8 + 2j ; j>=4: K = 16 + kh*8 + 2(j-4) (same as A).
DEV v16bf load_bt_frag(const unsigned short* base, int ldb, int lane) {
  const int n = lane & 15, kh = lane >> 4;
  FragU r;
#pragma unroll
  for (int j = 0; j < 8; ++j) {
    const int k0 = (j < 4) ? (kh * 8 + 2 * j) : (16 + kh * 8 + 2 * (j - 4));
    r.u[j] = *(const unsigned int*)(base + n * ldb + k0);
  }
  return r.v;
}

// ---------------------------------------------------------------------------
// GEMM: C[M,N] = A[M,K](bf16) * B[K,N](bf16 given as Bt[N,K] row-major).
// EPI: 0 f32 store | 1 f32 residual add | 2 bf16 store | 3 bf16 GELU store
// Block tile 128x128, BK=32, 256 threads = 8 waves in 4(M)x2(N), wave 32x64.
// Double-buffered LDS; staging via async global->LDS b64 when available.
// M multiple of 128; K multiple of 32; N guarded (OOB B rows clamp — their
// products land only in unstored columns).
// ---------------------------------------------------------------------------
template <int EPI>
__global__ __launch_bounds__(256)
void gemm_bf16_kernel(const unsigned short* __restrict__ A,
                      const unsigned short* __restrict__ Bt,
                      float* __restrict__ outF,
                      unsigned short* __restrict__ outB,
                      const float* __restrict__ res,
                      int M, int N, int K) {
  __shared__ alignas(16) unsigned short As[2][128 * LDA];
  __shared__ alignas(16) unsigned short Bs[2][128 * LDA];

  const int tid = threadIdx.x, lane = tid & 31, wid = tid >> 5;
  const int wm = wid >> 1, wn = wid & 1;
  const int bm = blockIdx.y * 128, bn = blockIdx.x * 128;

  v8f acc[2][4];
#pragma unroll
  for (int i = 0; i < 2; ++i)
#pragma unroll
    for (int j = 0; j < 4; ++j) acc[i][j] = v8f_zero();

  const int nk = K >> 5;

  auto stage = [&](int t, int buf) {
    const int k0 = t << 5;
#pragma unroll
    for (int i = 0; i < 4; ++i) {
      const int idx = tid + i * 256;
      const int row = idx >> 3, c4 = (idx & 7) * 4;
      const int rowB = (bn + row < N) ? (bn + row) : (N - 1);
      async_copy_b64(A  + (size_t)(bm + row) * K + k0 + c4, &As[buf][row * LDA + c4]);
      async_copy_b64(Bt + (size_t)rowB       * K + k0 + c4, &Bs[buf][row * LDA + c4]);
    }
  };

  stage(0, 0);
  async_wait_all();
  __syncthreads();
  for (int t = 0; t < nk; ++t) {
    const int cur = t & 1;
    if (t + 1 < nk) stage(t + 1, cur ^ 1);

    v16bf af[2], bfv[4];
#pragma unroll
    for (int i = 0; i < 2; ++i)
      af[i] = load_a_frag(&As[cur][(wm * 32 + i * 16) * LDA], LDA, lane);
#pragma unroll
    for (int j = 0; j < 4; ++j)
      bfv[j] = load_bt_frag(&Bs[cur][(wn * 64 + j * 16) * LDA], LDA, lane);
#pragma unroll
    for (int i = 0; i < 2; ++i)
#pragma unroll
      for (int j = 0; j < 4; ++j)
        acc[i][j] = __builtin_amdgcn_wmma_f32_16x16x32_bf16(
            false, af[i], false, bfv[j], (short)0, acc[i][j], false, false);

    async_wait_all();
    __syncthreads();
  }

  const int hlf = lane >> 4, nl = lane & 15;
#pragma unroll
  for (int i = 0; i < 2; ++i)
#pragma unroll
    for (int j = 0; j < 4; ++j)
#pragma unroll
      for (int r = 0; r < 8; ++r) {
        const int m = bm + wm * 32 + i * 16 + r + 8 * hlf;
        const int n = bn + wn * 64 + j * 16 + nl;
        if (n < N) {
          const size_t o = (size_t)m * N + n;
          const float val = acc[i][j][r];
          if constexpr (EPI == 0) {
            outF[o] = val;
          } else if constexpr (EPI == 1) {
            outF[o] = res[o] + val;
          } else if constexpr (EPI == 2) {
            outB[o] = f2bf(val);
          } else {
            const float g = 0.5f * val * (1.f + erff(val * 0.70710678118654752f));
            outB[o] = f2bf(g);
          }
        }
      }
}

// ---------------------------------------------------------------------------
// Sliding-window causal attention, flash-style online softmax.
// qkv: bf16 [BT, 3D] (q | k | v per row). y: bf16 [BT, D].
// One block (128 thr = 4 waves) per (b, h, 64-query block); wave owns 16
// queries. Key chunks of 64 cover [max(0, qs-W), qs+63]. Only the causal-edge
// chunk (kb==qb) and the window-edge chunk (kb==cstart when qs>=W) need the
// per-element mask — interior chunks skip all compares (uniform branch).
// LDS strides padded to 74 (37 dword banks, coprime with 64).
// ---------------------------------------------------------------------------
__global__ __launch_bounds__(128)
void attn_kernel(const unsigned short* __restrict__ qkv,
                 unsigned short* __restrict__ y) {
  __shared__ alignas(16) unsigned short Qs[64 * 74];       // [q][d]
  __shared__ alignas(16) unsigned short Ks[64 * 74];       // [key][d]
  __shared__ alignas(16) unsigned short Vt[64 * 74];       // [d][key]
  __shared__ alignas(16) unsigned short Ps[4 * 16 * 74];   // per-wave P [16][64]

  const int qb  = blockIdx.x & 15;
  const int hh  = (blockIdx.x >> 4) % Hh;
  const int b   = blockIdx.x / (16 * Hh);
  const int tid = threadIdx.x, lane = tid & 31, wv = tid >> 5;
  const int hlf = lane >> 4, nl = lane & 15;
  const int qs  = qb * 64;
  const size_t rowbase = (size_t)(b * Tt + qs);

  // Stage Q tile 64x64
#pragma unroll
  for (int i = 0; i < 8; ++i) {
    const int idx = tid + i * 128;
    const int row = idx >> 4, c4 = (idx & 15) * 4;
    const uint2 v = *(const uint2*)(qkv + (rowbase + row) * D3 + hh * 64 + c4);
    unsigned short* dst = Qs + row * 74 + c4;
    *(unsigned int*)(dst)     = v.x;
    *(unsigned int*)(dst + 2) = v.y;
  }
  __syncthreads();

  const v16bf qa0 = load_a_frag(Qs + (wv * 16) * 74, 74, lane);       // d 0..31
  const v16bf qa1 = load_a_frag(Qs + (wv * 16) * 74 + 32, 74, lane);  // d 32..63

  v8f o[4];
  float mrow[8], lrow[8];
#pragma unroll
  for (int r = 0; r < 8; ++r) { mrow[r] = -1e30f; lrow[r] = 0.f; }
#pragma unroll
  for (int dt = 0; dt < 4; ++dt) o[dt] = v8f_zero();

  const int cstart = (qs >= Ww) ? ((qs - Ww) >> 6) : 0;
  for (int kb = cstart; kb <= qb; ++kb) {
    __syncthreads();
    // Stage K chunk (row-major copy) and V chunk (transposed scatter)
#pragma unroll
    for (int i = 0; i < 8; ++i) {
      const int idx = tid + i * 128;
      const int key = idx >> 4, c4 = (idx & 15) * 4;
      const size_t krow = (size_t)(b * Tt + kb * 64 + key);
      {
        const uint2 v = *(const uint2*)(qkv + krow * D3 + Dd + hh * 64 + c4);
        unsigned short* dst = Ks + key * 74 + c4;
        *(unsigned int*)(dst)     = v.x;
        *(unsigned int*)(dst + 2) = v.y;
      }
      {
        const uint2 v = *(const uint2*)(qkv + krow * D3 + 2 * Dd + hh * 64 + c4);
        const unsigned short* pv = (const unsigned short*)&v;
#pragma unroll
        for (int e = 0; e < 4; ++e) Vt[(c4 + e) * 74 + key] = pv[e];
      }
    }
    __syncthreads();

    const bool edge = (kb == qb) || ((qs >= Ww) && (kb == cstart));

    // Scores S = (Q K^T) * scale (+ mask on edge chunks); chunk row-max
    v8f s[4];
    float cm[8];
#pragma unroll
    for (int r = 0; r < 8; ++r) cm[r] = -1e30f;
#pragma unroll
    for (int nt = 0; nt < 4; ++nt) {
      const v16bf kb0 = load_bt_frag(Ks + (nt * 16) * 74, 74, lane);       // d 0..31
      const v16bf kb1 = load_bt_frag(Ks + (nt * 16) * 74 + 32, 74, lane);  // d 32..63
      v8f t = v8f_zero();
      t = __builtin_amdgcn_wmma_f32_16x16x32_bf16(false, qa0, false, kb0, (short)0, t, false, false);
      t = __builtin_amdgcn_wmma_f32_16x16x32_bf16(false, qa1, false, kb1, (short)0, t, false, false);
      if (edge) {
#pragma unroll
        for (int r = 0; r < 8; ++r) {
          const int q  = qs + wv * 16 + r + 8 * hlf;
          const int ky = kb * 64 + nt * 16 + nl;
          float val = t[r] * 0.125f;                     // 1/sqrt(64)
          if (ky > q || ky < q - Ww) val = -1e30f;
          t[r] = val;
          cm[r] = fmaxf(cm[r], val);
        }
      } else {
#pragma unroll
        for (int r = 0; r < 8; ++r) {
          const float val = t[r] * 0.125f;
          t[r] = val;
          cm[r] = fmaxf(cm[r], val);
        }
      }
      s[nt] = t;
    }
    // Row max across the 16 lanes of each half
#pragma unroll
    for (int r = 0; r < 8; ++r) {
#pragma unroll
      for (int msk = 1; msk <= 8; msk <<= 1)
        cm[r] = fmaxf(cm[r], __shfl_xor(cm[r], msk, 32));
    }
    float alpha[8], rs[8];
#pragma unroll
    for (int r = 0; r < 8; ++r) {
      const float mn = fmaxf(mrow[r], cm[r]);
      alpha[r] = __expf(mrow[r] - mn);
      mrow[r] = mn;
      rs[r] = 0.f;
    }
    // P = exp(S - m); stage bf16 P to LDS (wave-private region)
#pragma unroll
    for (int nt = 0; nt < 4; ++nt)
#pragma unroll
      for (int r = 0; r < 8; ++r) {
        const float p = __expf(s[nt][r] - mrow[r]);
        rs[r] += p;
        Ps[wv * (16 * 74) + (r + 8 * hlf) * 74 + nt * 16 + nl] = f2bf(p);
      }
#pragma unroll
    for (int r = 0; r < 8; ++r) {
#pragma unroll
      for (int msk = 1; msk <= 8; msk <<= 1)
        rs[r] += __shfl_xor(rs[r], msk, 32);
      lrow[r] = lrow[r] * alpha[r] + rs[r];
    }
    __syncthreads();   // make P visible cross-lane before re-fragmenting

    const v16bf pa0 = load_a_frag(Ps + wv * (16 * 74), 74, lane);       // keys 0..31
    const v16bf pa1 = load_a_frag(Ps + wv * (16 * 74) + 32, 74, lane);  // keys 32..63
#pragma unroll
    for (int dt = 0; dt < 4; ++dt) {
      const v16bf vb0 = load_bt_frag(Vt + (dt * 16) * 74, 74, lane);       // keys 0..31
      const v16bf vb1 = load_bt_frag(Vt + (dt * 16) * 74 + 32, 74, lane);  // keys 32..63
      v8f t = o[dt];
#pragma unroll
      for (int r = 0; r < 8; ++r) t[r] *= alpha[r];
      t = __builtin_amdgcn_wmma_f32_16x16x32_bf16(false, pa0, false, vb0, (short)0, t, false, false);
      t = __builtin_amdgcn_wmma_f32_16x16x32_bf16(false, pa1, false, vb1, (short)0, t, false, false);
      o[dt] = t;
    }
  }

  // y = O / l  (bf16)
#pragma unroll
  for (int dt = 0; dt < 4; ++dt)
#pragma unroll
    for (int r = 0; r < 8; ++r) {
      const size_t row = rowbase + wv * 16 + r + 8 * hlf;
      const int col = hh * 64 + dt * 16 + nl;
      y[row * Dd + col] = f2bf(o[dt][r] / lrow[r]);
    }
}

// ---------------------------------------------------------------------------
// LayerNorm: f32 [rows, 768] -> bf16 [rows, 768]; one 256-thread block per row.
// ---------------------------------------------------------------------------
__global__ __launch_bounds__(256)
void layernorm_bf16_kernel(const float* __restrict__ x, const float* __restrict__ g,
                           unsigned short* __restrict__ out) {
  __shared__ float red[256];
  const int row = blockIdx.x, tid = threadIdx.x;
  const float* xr = x + (size_t)row * Dd;
  float v[3];
  float sum = 0.f;
#pragma unroll
  for (int i = 0; i < 3; ++i) { v[i] = xr[tid + i * 256]; sum += v[i]; }
  red[tid] = sum; __syncthreads();
  for (int st = 128; st > 0; st >>= 1) {
    if (tid < st) red[tid] += red[tid + st];
    __syncthreads();
  }
  const float mu = red[0] * (1.f / Dd);
  __syncthreads();
  float vs = 0.f;
#pragma unroll
  for (int i = 0; i < 3; ++i) { const float d = v[i] - mu; vs += d * d; }
  red[tid] = vs; __syncthreads();
  for (int st = 128; st > 0; st >>= 1) {
    if (tid < st) red[tid] += red[tid + st];
    __syncthreads();
  }
  const float rstd = rsqrtf(red[0] * (1.f / Dd) + 1e-5f);
#pragma unroll
  for (int i = 0; i < 3; ++i) {
    const int d = tid + i * 256;
    out[(size_t)row * Dd + d] = f2bf((v[i] - mu) * rstd * g[d]);
  }
}

__global__ __launch_bounds__(256)
void embed_kernel(const int* __restrict__ idx, const float* __restrict__ wte,
                  const float* __restrict__ wpe, float* __restrict__ x) {
  const int row = blockIdx.x, tid = threadIdx.x;
  const int t = row & (Tt - 1);
  const int tok = idx[row];
#pragma unroll
  for (int i = 0; i < 3; ++i) {
    const int d = tid + i * 256;
    x[(size_t)row * Dd + d] = wte[(size_t)tok * Dd + d] + wpe[(size_t)t * Dd + d];
  }
}

__global__ void cvt_bf16_kernel(const float* __restrict__ in,
                                unsigned short* __restrict__ out, size_t n) {
  size_t i = (size_t)blockIdx.x * blockDim.x + threadIdx.x;
  const size_t stride = (size_t)gridDim.x * blockDim.x;
  for (; i < n; i += stride) out[i] = f2bf(in[i]);
}

// Batched transpose+convert: in = L matrices [K,N] f32 -> out [N,K] bf16.
// LDS-tiled 32x32 so both global read and write are coalesced.
// grid: (N/32, K/32, L), block: 256 = 32 x 8.
__global__ __launch_bounds__(256)
void transpose_cvt_kernel(const float* __restrict__ in,
                          unsigned short* __restrict__ out, int K, int N) {
  __shared__ float tile[32][33];
  const int l = blockIdx.z;
  const int kt = blockIdx.y * 32, nt = blockIdx.x * 32;
  const float* src = in + (size_t)l * K * N;
  unsigned short* dst = out + (size_t)l * K * N;
  const int c = threadIdx.x & 31, r0 = threadIdx.x >> 5;
#pragma unroll
  for (int i = 0; i < 4; ++i) {
    const int r = r0 + i * 8;
    tile[r][c] = src[(size_t)(kt + r) * N + nt + c];
  }
  __syncthreads();
#pragma unroll
  for (int i = 0; i < 4; ++i) {
    const int r = r0 + i * 8;
    dst[(size_t)(nt + r) * K + kt + c] = f2bf(tile[c][r]);
  }
}

// ---------------------------------------------------------------------------
extern "C" void kernel_launch(void* const* d_in, const int* /*in_sizes*/, int /*n_in*/,
                              void* d_out, int /*out_size*/, void* d_ws, size_t /*ws_size*/,
                              hipStream_t stream) {
  const int*   idx    = (const int*)  d_in[0];
  const float* wte    = (const float*)d_in[1];
  const float* wpe    = (const float*)d_in[2];
  const float* ln1_g  = (const float*)d_in[3];
  const float* qkv_w  = (const float*)d_in[4];
  const float* proj_w = (const float*)d_in[5];
  const float* ln2_g  = (const float*)d_in[6];
  const float* fc_w   = (const float*)d_in[7];
  const float* fc2_w  = (const float*)d_in[8];
  const float* lnf_g  = (const float*)d_in[9];
  float* out = (float*)d_out;
  char* ws = (char*)d_ws;

  size_t off = 0;
  auto carve = [&](size_t bytes) -> void* {
    void* p = ws + off;
    off = (off + bytes + 255) & ~(size_t)255;
    return p;
  };
  // Transposed bf16 weights: [N,K] per layer
  unsigned short* wqkvT  = (unsigned short*)carve((size_t)Ll * Dd * D3 * 2);
  unsigned short* wprojT = (unsigned short*)carve((size_t)Ll * Dd * Dd * 2);
  unsigned short* wfcT   = (unsigned short*)carve((size_t)Ll * Dd * D4 * 2);
  unsigned short* wfc2T  = (unsigned short*)carve((size_t)Ll * D4 * Dd * 2);
  unsigned short* wteBf  = (unsigned short*)carve((size_t)Vv * Dd * 2);  // already [N=V, K=D]
  float*          x      = (float*)         carve((size_t)BT * Dd * 4);
  unsigned short* hbf    = (unsigned short*)carve((size_t)BT * Dd * 2);
  unsigned short* qkvbf  = (unsigned short*)carve((size_t)BT * D3 * 2);
  unsigned short* ybf    = (unsigned short*)carve((size_t)BT * Dd * 2);
  unsigned short* ubf    = (unsigned short*)carve((size_t)BT * D4 * 2);

  // Weight conversion (bf16) + transpose to [N,K] — all coalesced
  transpose_cvt_kernel<<<dim3(D3 / 32, Dd / 32, Ll), 256, 0, stream>>>(qkv_w,  wqkvT,  Dd, D3);
  transpose_cvt_kernel<<<dim3(Dd / 32, Dd / 32, Ll), 256, 0, stream>>>(proj_w, wprojT, Dd, Dd);
  transpose_cvt_kernel<<<dim3(D4 / 32, Dd / 32, Ll), 256, 0, stream>>>(fc_w,   wfcT,   Dd, D4);
  transpose_cvt_kernel<<<dim3(Dd / 32, D4 / 32, Ll), 256, 0, stream>>>(fc2_w,  wfc2T,  D4, Dd);
  cvt_bf16_kernel<<<8192, 256, 0, stream>>>(wte, wteBf, (size_t)Vv * Dd);

  embed_kernel<<<BT, 256, 0, stream>>>(idx, wte, wpe, x);

  const dim3 blk(256);
  for (int l = 0; l < Ll; ++l) {
    layernorm_bf16_kernel<<<BT, 256, 0, stream>>>(x, ln1_g + (size_t)l * Dd, hbf);
    gemm_bf16_kernel<2><<<dim3(D3 / 128, BT / 128), blk, 0, stream>>>(
        hbf, wqkvT + (size_t)l * Dd * D3, nullptr, qkvbf, nullptr, BT, D3, Dd);
    attn_kernel<<<Bb * Hh * (Tt / 64), 128, 0, stream>>>(qkvbf, ybf);
    gemm_bf16_kernel<1><<<dim3(Dd / 128, BT / 128), blk, 0, stream>>>(
        ybf, wprojT + (size_t)l * Dd * Dd, x, nullptr, x, BT, Dd, Dd);
    layernorm_bf16_kernel<<<BT, 256, 0, stream>>>(x, ln2_g + (size_t)l * Dd, hbf);
    gemm_bf16_kernel<3><<<dim3(D4 / 128, BT / 128), blk, 0, stream>>>(
        hbf, wfcT + (size_t)l * Dd * D4, nullptr, ubf, nullptr, BT, D4, Dd);
    gemm_bf16_kernel<1><<<dim3(Dd / 128, BT / 128), blk, 0, stream>>>(
        ubf, wfc2T + (size_t)l * D4 * Dd, x, nullptr, x, BT, Dd, D4);
  }

  layernorm_bf16_kernel<<<BT, 256, 0, stream>>>(x, lnf_g, hbf);
  gemm_bf16_kernel<0><<<dim3((Vv + 127) / 128, BT / 128), blk, 0, stream>>>(
      hbf, wteBf, out, nullptr, nullptr, BT, Vv, Dd);
}
